// MultiDense_80908593923415
// MI455X (gfx1250) — matrix operand order repaired and checked by reading, not compiled, not run on hardware
//
#include <hip/hip_runtime.h>

typedef __attribute__((ext_vector_type(16))) _Float16 v16h;
typedef __attribute__((ext_vector_type(8)))  _Float16 v8h;
typedef __attribute__((ext_vector_type(4)))  _Float16 v4h;
typedef __attribute__((ext_vector_type(8)))  float    v8f;
typedef __attribute__((ext_vector_type(4)))  float    v4f;

#define B_TOK   65536
#define L_EXP   16
#define I_DIM   256
#define O_DIM   256
#define KSTEPS  128          // 4096 / 32
#define BM      128
#define VSTRIDE 264          // 256 + 8 halves padding (bank-conflict relief)
#define LKSTRIDE 40          // 32 + 8 halves padding

// ---------------------------------------------------------------------------
// Prepack: W (4096x256 f32, row-major k,o) and bias (16x256 f32) -> f16
// B-fragments in workspace, laid out exactly as v_wmma_f32_16x16x32_f16 wants
// its B operand per lane:
//   tile t = kt*16 + nt, kt in [0,128], nt in [0,16)
//   lane: column n = lane&15, Kbase = (lane>=16)?8:0, element e -> K=Kbase+e (e<8), 16+Kbase+(e-8) (e>=8)
// Tile kt==128 is the bias K-extension (rows 0..15 = bias, 16..31 = 0).
// ---------------------------------------------------------------------------
__global__ __launch_bounds__(256) void prepack_kernel(
    const float* __restrict__ W, const float* __restrict__ bias,
    _Float16* __restrict__ ws)
{
  int idx  = blockIdx.x * 256 + threadIdx.x;   // [0, 129*16*32)
  int lane = idx & 31;
  int tile = idx >> 5;
  if (tile >= (KSTEPS + 1) * 16) return;
  int kt = tile >> 4, nt = tile & 15;
  int kbase = (lane >> 4) * 8;
  int o = nt * 16 + (lane & 15);
  v16h v;
  if (kt < KSTEPS) {
    int krow = kt * 32 + kbase;
    #pragma unroll
    for (int e = 0; e < 16; ++e)
      v[e] = (_Float16)W[(krow + e + ((e >> 3) << 3)) * O_DIM + o];
  } else {
    #pragma unroll
    for (int e = 0; e < 16; ++e) {
      int t = kbase + e + ((e >> 3) << 3);
      v[e] = (t < L_EXP) ? (_Float16)bias[t * O_DIM + o] : (_Float16)0.f;
    }
  }
  *(v16h*)&ws[(tile * 32 + lane) * 16] = v;
}

// ---------------------------------------------------------------------------
// Main GEMM: out[128 x 256] per block = A(on-the-fly f16) x Wf16 (+ bias ext).
// 8 waves (wave32) in a 2x4 grid; each wave computes 64x64 via 4x4 WMMA tiles.
// K loop is expert-outer (16) x 8 k-steps so routing scalars are hoisted.
// ---------------------------------------------------------------------------
__global__ __launch_bounds__(256) void md_kernel(
    const float* __restrict__ values, const float* __restrict__ lookups,
    const _Float16* __restrict__ wsB, float* __restrict__ out)
{
  __shared__ _Float16 smV[BM * VSTRIDE];   // values tile, f16
  __shared__ _Float16 smL[BM * LKSTRIDE];  // lookups + zero extension, f16

  const int tid  = threadIdx.x;
  const int row0 = blockIdx.x * BM;

  // ---- stage values -> f16 LDS (coalesced f32x4 loads, ds_store_b64) ----
  for (int i = tid; i < BM * 64; i += 256) {            // 128 rows * 64 float4
    int r = i >> 6, c = (i & 63) << 2;
    v4f f = *(const v4f*)&values[(row0 + r) * I_DIM + c];
    v4h h;
    h[0] = (_Float16)f[0]; h[1] = (_Float16)f[1];
    h[2] = (_Float16)f[2]; h[3] = (_Float16)f[3];
    *(v4h*)&smV[r * VSTRIDE + c] = h;
  }
  // ---- stage lookups + zeroed 16-col extension for the bias k-step ----
  for (int i = tid; i < BM * L_EXP; i += 256) {
    int m = i >> 4, t = i & 15;
    smL[m * LKSTRIDE + t]      = (_Float16)lookups[(row0 + m) * L_EXP + t];
    smL[m * LKSTRIDE + 16 + t] = (_Float16)0.f;
  }
  __syncthreads();

  const int wave  = tid >> 5, lane = tid & 31;
  const int wm    = wave >> 2, wn  = wave & 3;     // 2 x 4 wave grid
  const int mlan  = lane & 15, hsel = lane >> 4;
  const int mbase = wm * 64,  ntbase = wn * 4;

  v8f acc[4][4] = {};
  const v16h* Bt = (const v16h*)wsB;

  // per-lane row indices for the 4 m-tiles
  int mrow[4];
  #pragma unroll
  for (int jm = 0; jm < 4; ++jm) mrow[jm] = mbase + jm * 16 + mlan;

  #pragma unroll 1
  for (int l = 0; l < L_EXP; ++l) {
    // hoist routing scalars: constant for the next 8 k-steps
    _Float16 s[4];
    #pragma unroll
    for (int jm = 0; jm < 4; ++jm)
      s[jm] = smL[mrow[jm] * LKSTRIDE + l];

    #pragma unroll 2
    for (int ks = 0; ks < 8; ++ks) {
      const int kk = l * 8 + ks;
      const int i0 = ks << 5;

      v16h bf[4];
      #pragma unroll
      for (int jn = 0; jn < 4; ++jn)
        bf[jn] = Bt[(kk * 16 + ntbase + jn) * 32 + lane];

      // double-buffered A fragments: disjoint register sets so the 4
      // co-exec WAR slots after each WMMA are filled with real work
      v16h af[2][4];
      #pragma unroll
      for (int jm = 0; jm < 4; ++jm) {
        int ih = i0 + (hsel << 3);
        v8h lo = *(const v8h*)&smV[mrow[jm] * VSTRIDE + ih];
        v8h hi = *(const v8h*)&smV[mrow[jm] * VSTRIDE + ih + 16];
        v16h a = __builtin_shufflevector(lo, hi,
                 0,1,2,3,4,5,6,7,8,9,10,11,12,13,14,15);
        af[ks & 1][jm] = a * s[jm];
      }

      #pragma unroll
      for (int jm = 0; jm < 4; ++jm)
        #pragma unroll
        for (int jn = 0; jn < 4; ++jn)
          acc[jm][jn] = __builtin_amdgcn_wmma_f32_16x16x32_f16(
              false, af[ks & 1][jm], false, bf[jn], (short)0,
              acc[jm][jn], false, false);
    }
  }

  // ---- peeled k-step: bias via the K-extension (A = lookups|0, B = bias|0)
  {
    v16h bf[4];
    #pragma unroll
    for (int jn = 0; jn < 4; ++jn)
      bf[jn] = Bt[(KSTEPS * 16 + ntbase + jn) * 32 + lane];

    v16h af[4];
    #pragma unroll
    for (int jm = 0; jm < 4; ++jm) {
      int t0 = hsel << 3;
      v8h lo = *(const v8h*)&smL[mrow[jm] * LKSTRIDE + t0];
      v8h hi = *(const v8h*)&smL[mrow[jm] * LKSTRIDE + t0 + 16];
      af[jm] = __builtin_shufflevector(lo, hi,
               0,1,2,3,4,5,6,7,8,9,10,11,12,13,14,15);
    }
    #pragma unroll
    for (int jm = 0; jm < 4; ++jm)
      #pragma unroll
      for (int jn = 0; jn < 4; ++jn)
        acc[jm][jn] = __builtin_amdgcn_wmma_f32_16x16x32_f16(
            false, af[jm], false, bf[jn], (short)0, acc[jm][jn], false, false);
  }

  // ---- epilogue: f32 stores (VGPR r -> row r + 8*hsel, col = lane&15) ----
  #pragma unroll
  for (int jm = 0; jm < 4; ++jm) {
    #pragma unroll
    for (int jn = 0; jn < 4; ++jn) {
      int o = wn * 64 + jn * 16 + mlan;
      #pragma unroll
      for (int r = 0; r < 8; ++r) {
        int m = mrow[jm] - mlan + r + (hsel << 3);
        out[(row0 + m) * O_DIM + o] = acc[jm][jn][r];
      }
    }
  }
}

// ---------------------------------------------------------------------------
extern "C" void kernel_launch(void* const* d_in, const int* in_sizes, int n_in,
                              void* d_out, int out_size, void* d_ws, size_t ws_size,
                              hipStream_t stream)
{
  const float* lookups = (const float*)d_in[0];   // (B, L)
  const float* values  = (const float*)d_in[1];   // (B, I)
  const float* W       = (const float*)d_in[2];   // (L, I, O) == (4096, 256)
  const float* bias    = (const float*)d_in[3];   // (L, O)
  float*       out     = (float*)d_out;           // (B, O)
  _Float16*    ws      = (_Float16*)d_ws;         // ~2.02 MB swizzled f16 W

  int total = (KSTEPS + 1) * 16 * 32;             // one thread per (tile,lane)
  prepack_kernel<<<(total + 255) / 256, 256, 0, stream>>>(W, bias, ws);
  md_kernel<<<B_TOK / BM, 256, 0, stream>>>(values, lookups, ws, out);
}
